// GraphSAGE_34248069219261
// MI455X (gfx1250) — matrix operand order repaired and checked by reading, not compiled, not run on hardware
//
#include <hip/hip_runtime.h>
#include <math.h>

// ---------------------------------------------------------------------------
// GraphSAGE (2-layer, mean aggr) for MI455X / gfx1250, wave32 + WMMA bf16.
//   h0 = relu( (agg0/deg) @ Wl0 + b0 + X  @ Wr0 )
//   h1 = relu( (agg1/deg) @ Wl1 + b1 + h0 @ Wr1 )
//   out = sigmoid( h1 @ Wout + bout )
// Aggregation = L2-resident f32 atomics (51MB node matrix fits in 192MB L2);
// dense math = v_wmma_f32_16x16x32_bf16 with LDS-staged A fragments.
// ---------------------------------------------------------------------------

#define IN_DIM 128
#define HIDDEN 128
#define OUT_DIM 64

typedef __attribute__((ext_vector_type(16))) __bf16 v16bf;
typedef __attribute__((ext_vector_type(8)))  float  v8f;

__device__ __forceinline__ __bf16 tobf(float f) { return (__bf16)f; }

// ------------------------- weight pre-pack (f32 -> bf16, B-operand layout) --
// Packed half index p = ((nt*KT + kt)*32 + lane)*16 + i  maps to
//   k = kt*32 + (i>>3)*16 + (lane>>4)*8 + (i&7),  n = nt*16 + (lane&15)
// so each lane's B fragment is one contiguous 32-byte load.
__global__ void pack_w(const float* __restrict__ W, __bf16* __restrict__ Wp,
                       int K, int N) {
  int p = blockIdx.x * 256 + threadIdx.x;
  if (p >= K * N) return;
  int i    = p & 15;
  int lane = (p >> 4) & 31;
  int tile = p >> 9;
  int KT   = K >> 5;
  int kt   = tile % KT;
  int nt   = tile / KT;
  int k = kt * 32 + (i >> 3) * 16 + (lane >> 4) * 8 + (i & 7);
  int n = nt * 16 + (lane & 15);
  Wp[p] = tobf(W[(long long)k * N + n]);
}

// ----------------------------------------------------------------- zeroing --
__global__ void zero_f32(float* __restrict__ p, long long n) {
  long long i = (long long)blockIdx.x * 256 + threadIdx.x;
  long long stride = (long long)gridDim.x * 256;
  for (; i < n; i += stride) p[i] = 0.0f;
}

// ------------------------------------------------------------ degree count --
__global__ void degree_k(const int* __restrict__ dst, float* __restrict__ deg, int E) {
  int e = blockIdx.x * 256 + threadIdx.x;
  if (e < E) atomicAdd(&deg[dst[e]], 1.0f);
}

// ---------------------------------------------------- edge scatter-add f32 --
// One wave per edge; each of 32 lanes moves 4 contiguous floats (float4 load,
// 4 global_atomic_add_f32). Feature rows are 512B -> full-cacheline locality.
__global__ void scatter_k(const float* __restrict__ X, const int* __restrict__ src,
                          const int* __restrict__ dst, float* __restrict__ agg, int E) {
  long long t = (long long)blockIdx.x * 256 + threadIdx.x;
  int e = (int)(t >> 5);
  if (e >= E) return;
  int lane = (int)(t & 31);
  long long s = src[e], d = dst[e];
  float4 v = *(const float4*)(X + s * IN_DIM + lane * 4);
  float* a = agg + d * IN_DIM + lane * 4;
  atomicAdd(a + 0, v.x);
  atomicAdd(a + 1, v.y);
  atomicAdd(a + 2, v.z);
  atomicAdd(a + 3, v.w);
}

// Convert one A-fragment entry (kt,lane2) of a 16x128 f32 tile into LDS in
// the CDNA5 16-bit A 16x32 layout: row = lane2&15, g = lane2>>4,
//   halves 0..7  : K = kt*32 + g*8 + i
//   halves 8..15 : K = kt*32 + 16 + g*8 + i
__device__ __forceinline__ void stage_entry(const float* __restrict__ M, int m0,
                                            int e, float scaleIn, const float* deg,
                                            __bf16* __restrict__ s) {
  int kt = e >> 5, lane2 = e & 31;
  int r  = m0 + (lane2 & 15);
  int kb = kt * 32 + (lane2 >> 4) * 8;
  float sc = (deg != nullptr) ? (1.0f / fmaxf(deg[r], 1.0f)) : scaleIn;
  const float* p = M + (long long)r * HIDDEN + kb;
  float4 a0 = *(const float4*)(p);
  float4 a1 = *(const float4*)(p + 4);
  float4 a2 = *(const float4*)(p + 16);
  float4 a3 = *(const float4*)(p + 20);
  __bf16* d = s + e * 16;
  d[0]  = tobf(a0.x * sc); d[1]  = tobf(a0.y * sc);
  d[2]  = tobf(a0.z * sc); d[3]  = tobf(a0.w * sc);
  d[4]  = tobf(a1.x * sc); d[5]  = tobf(a1.y * sc);
  d[6]  = tobf(a1.z * sc); d[7]  = tobf(a1.w * sc);
  d[8]  = tobf(a2.x * sc); d[9]  = tobf(a2.y * sc);
  d[10] = tobf(a2.z * sc); d[11] = tobf(a2.w * sc);
  d[12] = tobf(a3.x * sc); d[13] = tobf(a3.y * sc);
  d[14] = tobf(a3.z * sc); d[15] = tobf(a3.w * sc);
}

// --------------------------------------------------- fused SAGE layer GEMM --
// Block = 256 threads = 8 waves; block owns 16 node rows; wave w = column
// tile w (16 of 128 outputs). A tiles (X and AGG/deg) are converted to bf16
// ONCE per block into LDS in fragment layout; K=128 in 4 steps, 2 WMMAs/step.
// OUT may alias X (layer-1 in place): all global reads of X feed LDS stores
// that complete before __syncthreads(); stores happen after; blocks own
// disjoint row ranges.
__global__ void __launch_bounds__(256)
sage_layer(const float* __restrict__ X, const float* __restrict__ AGG,
           const float* __restrict__ DEG,
           const __bf16* __restrict__ WLp, const __bf16* __restrict__ WRp,
           const float* __restrict__ BIAS, float* __restrict__ OUT) {
  __shared__ __align__(32) __bf16 sx[4 * 32 * 16];   // X tile, frag layout (4KB)
  __shared__ __align__(32) __bf16 sa[4 * 32 * 16];   // AGG/deg tile      (4KB)

  const int lane = threadIdx.x & 31;
  const int nt   = threadIdx.x >> 5;                 // column tile 0..7
  const int g    = lane >> 4;
  const int m0   = blockIdx.x * 16;

  // Cooperative stage: threads 0..127 -> X entries, 128..255 -> AGG entries.
  {
    int t = threadIdx.x;
    int e = t & 127;
    if (t < 128) stage_entry(X,   m0, e, 1.0f, nullptr, sx);
    else         stage_entry(AGG, m0, e, 1.0f, DEG,     sa);
  }
  __syncthreads();

  const v16bf* wl = (const v16bf*)WLp;
  const v16bf* wr = (const v16bf*)WRp;
  v8f acc = {};
#pragma unroll
  for (int kt = 0; kt < 4; ++kt) {
    v16bf av = *(const v16bf*)(sa + (kt * 32 + lane) * 16);
    v16bf xv = *(const v16bf*)(sx + (kt * 32 + lane) * 16);
    v16bf bl = wl[(nt * 4 + kt) * 32 + lane];
    v16bf br = wr[(nt * 4 + kt) * 32 + lane];
    acc = __builtin_amdgcn_wmma_f32_16x16x32_bf16(false, av, false, bl,
                                                  (short)0, acc, false, false);
    acc = __builtin_amdgcn_wmma_f32_16x16x32_bf16(false, xv, false, br,
                                                  (short)0, acc, false, false);
  }

  const int col = nt * 16 + (lane & 15);
  const float b = BIAS[col];
#pragma unroll
  for (int j = 0; j < 8; ++j) {
    float v = acc[j] + b;                            // D row = j + 8*g
    OUT[(long long)(m0 + j + 8 * g) * HIDDEN + col] = fmaxf(v, 0.0f);
  }
}

// -------------------------------------------------------- output head GEMM --
__global__ void __launch_bounds__(128)
out_layer(const float* __restrict__ H, const __bf16* __restrict__ WOp,
          const float* __restrict__ BIAS, float* __restrict__ OUT) {
  __shared__ __align__(32) __bf16 sh[4 * 32 * 16];   // H tile, frag layout (4KB)

  const int lane = threadIdx.x & 31;
  const int nt   = threadIdx.x >> 5;                 // column tile 0..3
  const int g    = lane >> 4;
  const int m0   = blockIdx.x * 16;

  stage_entry(H, m0, threadIdx.x & 127, 1.0f, nullptr, sh);
  __syncthreads();

  const v16bf* wo = (const v16bf*)WOp;
  v8f acc = {};
#pragma unroll
  for (int kt = 0; kt < 4; ++kt) {
    v16bf hv = *(const v16bf*)(sh + (kt * 32 + lane) * 16);
    v16bf bw = wo[(nt * 4 + kt) * 32 + lane];
    acc = __builtin_amdgcn_wmma_f32_16x16x32_bf16(false, hv, false, bw,
                                                  (short)0, acc, false, false);
  }

  const int col = nt * 16 + (lane & 15);
  const float b = BIAS[col];
#pragma unroll
  for (int j = 0; j < 8; ++j) {
    float v = acc[j] + b;
    OUT[(long long)(m0 + j + 8 * g) * OUT_DIM + col] = 1.0f / (1.0f + expf(-v));
  }
}

// ---------------------------------------------------------------------------
extern "C" void kernel_launch(void* const* d_in, const int* in_sizes, int n_in,
                              void* d_out, int out_size, void* d_ws, size_t ws_size,
                              hipStream_t stream) {
  const float* x    = (const float*)d_in[0];
  const int*   ei   = (const int*)d_in[1];      // [2, E] (src row, dst row)
  const float* wl0  = (const float*)d_in[2];
  const float* bl0  = (const float*)d_in[3];
  const float* wr0  = (const float*)d_in[4];
  const float* wl1  = (const float*)d_in[5];
  const float* bl1  = (const float*)d_in[6];
  const float* wr1  = (const float*)d_in[7];
  const float* wout = (const float*)d_in[8];
  const float* bout = (const float*)d_in[9];

  const int NN = in_sizes[0] / IN_DIM;          // 100000 nodes
  const int E  = in_sizes[1] / 2;               // 1600000 edges
  const int* src = ei;
  const int* dst = ei + E;

  // Workspace layout (all offsets 16B-aligned): deg | agg | h | packed weights
  float* deg = (float*)d_ws;
  float* agg = deg + NN;
  float* h   = agg + (size_t)NN * HIDDEN;
  __bf16* wl0p = (__bf16*)(h + (size_t)NN * HIDDEN);
  __bf16* wr0p = wl0p + HIDDEN * HIDDEN;
  __bf16* wl1p = wr0p + HIDDEN * HIDDEN;
  __bf16* wr1p = wl1p + HIDDEN * HIDDEN;
  __bf16* wop  = wr1p + HIDDEN * HIDDEN;

  const int rowTiles = NN / 16;                 // 6250

  // Pack weights to bf16 in WMMA B-operand layout (once per call).
  pack_w<<<(HIDDEN * HIDDEN + 255) / 256, 256, 0, stream>>>(wl0, wl0p, HIDDEN, HIDDEN);
  pack_w<<<(HIDDEN * HIDDEN + 255) / 256, 256, 0, stream>>>(wr0, wr0p, HIDDEN, HIDDEN);
  pack_w<<<(HIDDEN * HIDDEN + 255) / 256, 256, 0, stream>>>(wl1, wl1p, HIDDEN, HIDDEN);
  pack_w<<<(HIDDEN * HIDDEN + 255) / 256, 256, 0, stream>>>(wr1, wr1p, HIDDEN, HIDDEN);
  pack_w<<<(HIDDEN * OUT_DIM + 255) / 256, 256, 0, stream>>>(wout, wop, HIDDEN, OUT_DIM);

  // Degree (shared by both layers).
  zero_f32<<<2048, 256, 0, stream>>>(deg, NN);
  degree_k<<<(E + 255) / 256, 256, 0, stream>>>(dst, deg, E);

  // Layer 0: aggregate x, then fused GEMM -> h (relu).
  zero_f32<<<4096, 256, 0, stream>>>(agg, (long long)NN * HIDDEN);
  scatter_k<<<(int)(((long long)E * 32 + 255) / 256), 256, 0, stream>>>(x, src, dst, agg, E);
  sage_layer<<<rowTiles, 256, 0, stream>>>(x, agg, deg, wl0p, wr0p, bl0, h);

  // Layer 1: aggregate h, fused GEMM in place -> h (relu).
  zero_f32<<<4096, 256, 0, stream>>>(agg, (long long)NN * HIDDEN);
  scatter_k<<<(int)(((long long)E * 32 + 255) / 256), 256, 0, stream>>>(h, src, dst, agg, E);
  sage_layer<<<rowTiles, 256, 0, stream>>>(h, agg, deg, wl1p, wr1p, bl1, h);

  // Output head: sigmoid(h @ Wout + bout) -> d_out.
  out_layer<<<rowTiles, 128, 0, stream>>>(h, wop, bout, (float*)d_out);
}
